// MultiHeadAttention_20512763806205
// MI455X (gfx1250) — matrix operand order repaired and checked by reading, not compiled, not run on hardware
//
#include <hip/hip_runtime.h>
#include <hip/hip_bf16.h>

typedef __bf16 bf16;
typedef __attribute__((ext_vector_type(16))) __bf16 v16bf;
typedef __attribute__((ext_vector_type(8)))  float  v8f;

#define DM   768
#define SEQ  2048
#define NB   2
#define NH   12
#define DK   64
#define MROWS (NB*SEQ)   // 4096
#define L2E  1.44269504f

union V16U { v16bf v; uint4 q[2]; };
union H8U  { bf16 h[8]; uint4 q; };

// ---------------------------------------------------------------------------
// Kernel 0: transpose + convert weights fp32 [K][N] -> bf16 [N][K]
// ---------------------------------------------------------------------------
__global__ __launch_bounds__(256) void prep_w(
    const float* __restrict__ W0, const float* __restrict__ W1,
    const float* __restrict__ W2, const float* __restrict__ W3,
    bf16* __restrict__ O0, bf16* __restrict__ O1,
    bf16* __restrict__ O2, bf16* __restrict__ O3)
{
    const float* W; bf16* O;
    switch (blockIdx.y) {
        case 0:  W = W0; O = O0; break;
        case 1:  W = W1; O = O1; break;
        case 2:  W = W2; O = O2; break;
        default: W = W3; O = O3; break;
    }
    int idx = blockIdx.x * 256 + threadIdx.x;   // over DM*DM
    int n = idx / DM, k = idx % DM;
    O[(size_t)n * DM + k] = (bf16)W[(size_t)k * DM + n];
}

// ---------------------------------------------------------------------------
// GEMM: C[4096x768] = A[4096x768] * Bt^T (+bias)*scale   (Bt is [N][K] bf16)
// OMODE 0: bf16 row-major out (Q/K projections; scale folded) - transposed acc
// OMODE 1: bf16 per-head-transposed Vt[b][h][d][s] out      - natural acc
// OMODE 2: f32 row-major out (final projection)             - transposed acc
// ---------------------------------------------------------------------------
template<int OMODE, bool AF32>
__global__ __launch_bounds__(128) void gemm_wmma_k(
    const void* __restrict__ Ap, const bf16* __restrict__ Bt,
    const float* __restrict__ bias, void* __restrict__ Cp, float scale)
{
    constexpr int  K     = DM;
    constexpr int  LDA   = 40;              // padded LDS stride (bf16 units)
    constexpr bool TRANS = (OMODE != 1);    // compute C^T for packed stores
    __shared__ __align__(16) bf16 As[64 * LDA];
    __shared__ __align__(16) bf16 Bs[64 * LDA];

    const int tid  = threadIdx.x;
    const int lane = tid & 31;
    const int w    = tid >> 5;
    const int lh   = lane >> 4;
    const int l16  = lane & 15;

    const int m0 = blockIdx.x * 64;
    const int n0 = blockIdx.y * 64;
    const int wr = (w >> 1) * 32;
    const int wc = (w & 1) * 32;

    v8f acc[2][2] = {};

    for (int k0 = 0; k0 < K; k0 += 32) {
        // ---- stage A (64x32, fp32->bf16 if needed) and B (64x32 from Bt) ----
        #pragma unroll
        for (int it = 0; it < 2; ++it) {
            int cid = tid + it * 128;       // 256 chunks of 8 halves
            int row = cid >> 2;
            int c8  = (cid & 3) * 8;
            H8U t;
            if (AF32) {
                const float* ap = (const float*)Ap + (size_t)(m0 + row) * K + k0 + c8;
                float4 f0 = *(const float4*)(ap);
                float4 f1 = *(const float4*)(ap + 4);
                t.h[0] = (bf16)f0.x; t.h[1] = (bf16)f0.y; t.h[2] = (bf16)f0.z; t.h[3] = (bf16)f0.w;
                t.h[4] = (bf16)f1.x; t.h[5] = (bf16)f1.y; t.h[6] = (bf16)f1.z; t.h[7] = (bf16)f1.w;
            } else {
                const bf16* ap = (const bf16*)Ap + (size_t)(m0 + row) * K + k0 + c8;
                t.q = *(const uint4*)ap;
            }
            *(uint4*)&As[row * LDA + c8] = t.q;
            *(uint4*)&Bs[row * LDA + c8] =
                *(const uint4*)(Bt + (size_t)(n0 + row) * K + k0 + c8);
            if (k0 + 32 < K) {
                if (AF32)
                    __builtin_prefetch((const float*)Ap + (size_t)(m0 + row) * K + k0 + 32 + c8, 0, 1);
                __builtin_prefetch(Bt + (size_t)(n0 + row) * K + k0 + 32 + c8, 0, 1);
            }
        }
        __syncthreads();

        if (TRANS) {
            // C^T = Bt * A^T : A-frag from Bs rows (n), B-frag from As rows (m)
            V16U fa[2], fb[2];
            #pragma unroll
            for (int ni = 0; ni < 2; ++ni) {
                const bf16* p = &Bs[(wc + ni * 16 + l16) * LDA + lh * 8];
                fa[ni].q[0] = *(const uint4*)p;
                fa[ni].q[1] = *(const uint4*)(p + 16);
            }
            #pragma unroll
            for (int mi = 0; mi < 2; ++mi) {
                const bf16* p = &As[(wr + mi * 16 + l16) * LDA + lh * 8];
                fb[mi].q[0] = *(const uint4*)p;
                fb[mi].q[1] = *(const uint4*)(p + 16);
            }
            #pragma unroll
            for (int ni = 0; ni < 2; ++ni)
                #pragma unroll
                for (int mi = 0; mi < 2; ++mi)
                    acc[ni][mi] = __builtin_amdgcn_wmma_f32_16x16x32_bf16(
                        false, fa[ni].v, false, fb[mi].v, (short)0, acc[ni][mi], false, false);
        } else {
            V16U fa[2], fb[2];
            #pragma unroll
            for (int mi = 0; mi < 2; ++mi) {
                const bf16* p = &As[(wr + mi * 16 + l16) * LDA + lh * 8];
                fa[mi].q[0] = *(const uint4*)p;
                fa[mi].q[1] = *(const uint4*)(p + 16);
            }
            #pragma unroll
            for (int ni = 0; ni < 2; ++ni) {
                const bf16* p = &Bs[(wc + ni * 16 + l16) * LDA + lh * 8];
                fb[ni].q[0] = *(const uint4*)p;
                fb[ni].q[1] = *(const uint4*)(p + 16);
            }
            #pragma unroll
            for (int mi = 0; mi < 2; ++mi)
                #pragma unroll
                for (int ni = 0; ni < 2; ++ni)
                    acc[mi][ni] = __builtin_amdgcn_wmma_f32_16x16x32_bf16(
                        false, fa[mi].v, false, fb[ni].v, (short)0, acc[mi][ni], false, false);
        }
        __syncthreads();
    }

    // ---- epilogue ----
    if (TRANS) {
        // lane owns row m = (wr+mi*16+l16); 8 consecutive columns nb..nb+7
        #pragma unroll
        for (int mi = 0; mi < 2; ++mi) {
            #pragma unroll
            for (int ni = 0; ni < 2; ++ni) {
                int m  = m0 + wr + mi * 16 + l16;
                int nb = n0 + wc + ni * 16 + lh * 8;
                float4 b0 = *(const float4*)&bias[nb];
                float4 b1 = *(const float4*)&bias[nb + 4];
                float bv[8] = { b0.x, b0.y, b0.z, b0.w, b1.x, b1.y, b1.z, b1.w };
                if (OMODE == 0) {
                    H8U t;
                    #pragma unroll
                    for (int r = 0; r < 8; ++r)
                        t.h[r] = (bf16)((acc[ni][mi][r] + bv[r]) * scale);
                    *(uint4*)&((bf16*)Cp)[(size_t)m * DM + nb] = t.q;
                } else {
                    float* out = (float*)Cp + (size_t)m * DM + nb;
                    float4 o0, o1;
                    o0.x = acc[ni][mi][0] + bv[0]; o0.y = acc[ni][mi][1] + bv[1];
                    o0.z = acc[ni][mi][2] + bv[2]; o0.w = acc[ni][mi][3] + bv[3];
                    o1.x = acc[ni][mi][4] + bv[4]; o1.y = acc[ni][mi][5] + bv[5];
                    o1.z = acc[ni][mi][6] + bv[6]; o1.w = acc[ni][mi][7] + bv[7];
                    *(float4*)out       = o0;
                    *(float4*)(out + 4) = o1;
                }
            }
        }
    } else {
        // OMODE 1: Vt[b][h][d][s], 8 consecutive s per lane -> packed store
        #pragma unroll
        for (int mi = 0; mi < 2; ++mi) {
            #pragma unroll
            for (int ni = 0; ni < 2; ++ni) {
                int   n     = n0 + wc + ni * 16 + l16;
                float bv    = bias[n];
                int   mbase = m0 + wr + mi * 16 + lh * 8;
                int h = n / DK, d = n % DK;
                int bb = mbase / SEQ, s = mbase % SEQ;   // tile never crosses batch
                H8U t;
                #pragma unroll
                for (int r = 0; r < 8; ++r) t.h[r] = (bf16)(acc[mi][ni][r] + bv);
                bf16* out = (bf16*)Cp + (((size_t)(bb * NH + h) * DK + d) * SEQ + s);
                *(uint4*)out = t.q;
            }
        }
    }
}

// ---------------------------------------------------------------------------
// Flash attention, fully transposed formulation:
//   S^T = K * Q^T  (per-lane softmax stats for query q = lane%16, 1 shuffle)
//   O^T = V^T * P^T
// K/V tiles staged by GLOBAL_LOAD_ASYNC_TO_LDS_B128 into double-buffered LDS
// (no staging VGPRs; DMA for tile i+1 overlaps compute of tile i).
// ---------------------------------------------------------------------------
__global__ __launch_bounds__(128, 1) void attn_k(
    const bf16* __restrict__ Q, const bf16* __restrict__ Kb,
    const bf16* __restrict__ Vt, bf16* __restrict__ Cc)
{
    constexpr int LDT = 72;                 // padded LDS stride (bf16 units)
    __shared__ __align__(16) bf16 Kt[2][64 * LDT];
    __shared__ __align__(16) bf16 Vs[2][64 * LDT];
    __shared__ __align__(16) bf16 Pl[4][16 * LDT];

    const int tid = threadIdx.x, lane = tid & 31, w = tid >> 5;
    const int l16 = lane & 15, lh = lane >> 4;
    const int blk = blockIdx.x;
    const int qt  = blk & 31;
    const int h   = (blk >> 5) % NH;
    const int bb  = blk / (32 * NH);
    const int q0  = qt * 64;

    // resident Q fragments in B-layout (column q = l16, dk in AMD table order)
    V16U qb[2];
    {
        const bf16* qp = Q + ((size_t)(bb * SEQ + q0 + w * 16 + l16) * DM + h * DK);
        #pragma unroll
        for (int s = 0; s < 2; ++s) {
            qb[s].q[0] = *(const uint4*)(qp + s * 32 + lh * 8);
            qb[s].q[1] = *(const uint4*)(qp + s * 32 + lh * 8 + 16);
        }
    }

    const bf16* kbase = Kb + ((size_t)bb * SEQ) * DM + h * DK;
    const bf16* vbase = Vt + ((size_t)(bb * NH + h) * DK) * SEQ;

    // async-DMA one 64-key K/V tile into LDS buffer `buf`
    auto issue_tile = [&](int kt0, int buf) {
        #pragma unroll
        for (int it = 0; it < 4; ++it) {
            int cid = tid + it * 128;       // 512 chunks of 8 halves
            int row = cid >> 3;
            int c8  = (cid & 7) * 8;
            unsigned    kdst = (unsigned)(size_t)&Kt[buf][row * LDT + c8];
            unsigned    vdst = (unsigned)(size_t)&Vs[buf][row * LDT + c8];
            const bf16* kg   = kbase + (size_t)(kt0 + row) * DM + c8;
            const bf16* vg   = vbase + (size_t)row * SEQ + kt0 + c8;
            asm volatile("global_load_async_to_lds_b128 %0, %1, off"
                         :: "v"(kdst), "v"(kg) : "memory");
            asm volatile("global_load_async_to_lds_b128 %0, %1, off"
                         :: "v"(vdst), "v"(vg) : "memory");
        }
    };

    issue_tile(0, 0);

    float mrun = -3.0e38f, lrun = 0.f;
    v8f ot[4] = {};

    for (int i = 0; i < SEQ / 64; ++i) {
        const int kt0 = i * 64;
        const int cur = i & 1;

        asm volatile("s_wait_asynccnt 0x0" ::: "memory");  // own DMA done
        __syncthreads();                                    // all waves' DMA visible

        if (kt0 + 64 < SEQ) issue_tile(kt0 + 64, cur ^ 1);  // overlap next tile

        const bf16* KtC = &Kt[cur][0];
        const bf16* VsC = &Vs[cur][0];

        // ---- S^T = K * Q^T : 4 key m-tiles x 2 dk k-steps ----
        v8f st[4] = {};
        #pragma unroll
        for (int mt = 0; mt < 4; ++mt) {
            #pragma unroll
            for (int s = 0; s < 2; ++s) {
                V16U ka;
                const bf16* p = &KtC[(mt * 16 + l16) * LDT + s * 32 + lh * 8];
                ka.q[0] = *(const uint4*)p;
                ka.q[1] = *(const uint4*)(p + 16);
                st[mt] = __builtin_amdgcn_wmma_f32_16x16x32_bf16(
                    false, ka.v, false, qb[s].v, (short)0, st[mt], false, false);
            }
        }

        // ---- per-lane online softmax for query q = l16 ----
        float tmax = -3.0e38f;
        #pragma unroll
        for (int mt = 0; mt < 4; ++mt)
            #pragma unroll
            for (int r = 0; r < 8; ++r)
                tmax = fmaxf(tmax, st[mt][r]);
        tmax = fmaxf(tmax, __shfl_xor(tmax, 16, 32));   // merge key halves

        float mn    = fmaxf(mrun, tmax);
        float alpha = __builtin_amdgcn_exp2f((mrun - mn) * L2E);
        mrun = mn;
        float c  = mn * L2E;
        float rs = 0.f;
        #pragma unroll
        for (int mt = 0; mt < 4; ++mt)
            #pragma unroll
            for (int r = 0; r < 8; ++r) {
                float p = __builtin_amdgcn_exp2f(st[mt][r] * L2E - c);
                st[mt][r] = p;
                rs += p;
            }
        rs += __shfl_xor(rs, 16, 32);
        lrun = lrun * alpha + rs;
        #pragma unroll
        for (int nt = 0; nt < 4; ++nt)
            #pragma unroll
            for (int r = 0; r < 8; ++r)
                ot[nt][r] *= alpha;

        // ---- pack P into LDS as [q][key] (4 x 16B stores per lane) ----
        #pragma unroll
        for (int mt = 0; mt < 4; ++mt) {
            H8U t;
            #pragma unroll
            for (int r = 0; r < 8; ++r) t.h[r] = (bf16)st[mt][r];
            *(uint4*)&Pl[w][l16 * LDT + mt * 16 + lh * 8] = t.q;
        }
        asm volatile("s_wait_dscnt 0x0" ::: "memory");   // in-wave store->load

        // ---- O^T += V^T * P^T : 2 key k-steps x 4 d-tiles ----
        #pragma unroll
        for (int s = 0; s < 2; ++s) {
            V16U pb;
            const bf16* pp = &Pl[w][l16 * LDT + s * 32 + lh * 8];
            pb.q[0] = *(const uint4*)pp;
            pb.q[1] = *(const uint4*)(pp + 16);
            #pragma unroll
            for (int nt = 0; nt < 4; ++nt) {
                V16U va;
                const bf16* vp = &VsC[(nt * 16 + l16) * LDT + s * 32 + lh * 8];
                va.q[0] = *(const uint4*)vp;
                va.q[1] = *(const uint4*)(vp + 16);
                ot[nt] = __builtin_amdgcn_wmma_f32_16x16x32_bf16(
                    false, va.v, false, pb.v, (short)0, ot[nt], false, false);
            }
        }
        __syncthreads();                 // buffer free for DMA reuse
    }

    // ---- normalize, pack, store concat row (4 x 16B stores per lane) ----
    float inv = 1.0f / lrun;
    bf16* outp = Cc + ((size_t)(bb * SEQ + q0 + w * 16 + l16) * DM + h * DK);
    #pragma unroll
    for (int nt = 0; nt < 4; ++nt) {
        H8U t;
        #pragma unroll
        for (int r = 0; r < 8; ++r) t.h[r] = (bf16)(ot[nt][r] * inv);
        *(uint4*)&outp[nt * 16 + lh * 8] = t.q;
    }
}

// ---------------------------------------------------------------------------
extern "C" void kernel_launch(void* const* d_in, const int* in_sizes, int n_in,
                              void* d_out, int out_size, void* d_ws, size_t ws_size,
                              hipStream_t stream)
{
    const float* q  = (const float*)d_in[0];
    const float* k  = (const float*)d_in[1];
    const float* v  = (const float*)d_in[2];
    const float* Wq = (const float*)d_in[3];
    const float* bq = (const float*)d_in[4];
    const float* Wk = (const float*)d_in[5];
    const float* bk = (const float*)d_in[6];
    const float* Wv = (const float*)d_in[7];
    const float* bv = (const float*)d_in[8];
    const float* Wo = (const float*)d_in[9];
    const float* bo = (const float*)d_in[10];
    float* out = (float*)d_out;

    char* ws = (char*)d_ws;
    const size_t WSZ = (size_t)DM * DM * sizeof(bf16);
    const size_t XSZ = (size_t)MROWS * DM * sizeof(bf16);
    bf16* Wq_t = (bf16*)ws; ws += WSZ;
    bf16* Wk_t = (bf16*)ws; ws += WSZ;
    bf16* Wv_t = (bf16*)ws; ws += WSZ;
    bf16* Wo_t = (bf16*)ws; ws += WSZ;
    bf16* Qb   = (bf16*)ws; ws += XSZ;
    bf16* Kb   = (bf16*)ws; ws += XSZ;
    bf16* Vt   = (bf16*)ws; ws += XSZ;
    bf16* Cc   = (bf16*)ws; ws += XSZ;

    prep_w<<<dim3(DM * DM / 256, 4), 256, 0, stream>>>(Wq, Wk, Wv, Wo,
                                                       Wq_t, Wk_t, Wv_t, Wo_t);
    dim3 gg(MROWS / 64, DM / 64);
    gemm_wmma_k<0, true ><<<gg, 128, 0, stream>>>(q,  Wq_t, bq, Qb,  0.125f);
    gemm_wmma_k<0, true ><<<gg, 128, 0, stream>>>(k,  Wk_t, bk, Kb,  1.0f);
    gemm_wmma_k<1, true ><<<gg, 128, 0, stream>>>(v,  Wv_t, bv, Vt,  1.0f);
    attn_k<<<NB * NH * (SEQ / 64), 128, 0, stream>>>(Qb, Kb, Vt, Cc);
    gemm_wmma_k<2, false><<<gg, 128, 0, stream>>>(Cc, Wo_t, bo, out, 1.0f);
}
